// NT2NAttentionModel_53291954209025
// MI455X (gfx1250) — compile-verified, matrix operand order checked by reading, not run on hardware
//
#include <hip/hip_runtime.h>
#include <hip/hip_bf16.h>
#include <math.h>

// ---------------- problem constants ----------------
#define SEQ      200
#define BATCH    128
#define NT_DIM   300
#define T_DIM    1200
#define IN_DIM   1500          // NT_DIM + T_DIM
#define HID      1500
#define G4       6000          // 4*HID
#define KPAD     1536          // K padded to multiple of 32
#define KCH      48            // 1536/32 chunks
#define KCH2     96            // combined [x|h] K chunks (3072/32)
#define NT_TILES 375           // 6000/16
#define NPW      5             // n-tiles per block (375 = 75*5)
#define KSTAGE   8             // k-chunks staged per LDS stage
#define NSTAGE   (KCH2/KSTAGE) // 12 stages (stages 0..5 = x, 6..11 = h)
#define PRED     97
#define PRED_PAD 112
#define PNT      7             // 112/16
#define MROWS    (SEQ*BATCH)   // 25600

typedef __bf16 bf16;
typedef __attribute__((ext_vector_type(16))) __bf16 v16bf;
typedef __attribute__((ext_vector_type(8)))  float  v8f;

// ---------------- helpers ----------------
__device__ __forceinline__ float sigm(float x) { return 1.0f / (1.0f + __expf(-x)); }
__device__ __forceinline__ bf16 f2bf(float f) { return (bf16)f; }

// Load a 16x32 fragment per the CDNA5 16-bit A/B VGPR layout: lane
// (hi = lane>>4) holds elems [hi*8..hi*8+7] and [16+hi*8..16+hi*8+7] of its
// row (A, row-major) / column (B, column-major tile). p -> 32-elem span (64B).
__device__ __forceinline__ v16bf load_frag32(const bf16* p, int hi) {
  union { uint4 u[2]; v16bf v; } f;
  const uint4* q = (const uint4*)p;
  f.u[0] = q[hi];
  f.u[1] = q[2 + hi];
  return f.v;
}

// CDNA5 async DMA: copy 16B from global to LDS, tracked by ASYNCcnt.
__device__ __forceinline__ void async_copy_b128(unsigned lds_off, const void* gptr) {
  asm volatile("global_load_async_to_lds_b128 %0, %1, off"
               :: "v"(lds_off), "v"(gptr)
               : "memory");
}
__device__ __forceinline__ void wait_asynccnt_10() {
  asm volatile("s_wait_asynccnt 0xa" ::: "memory");
}
__device__ __forceinline__ void wait_asynccnt_0() {
  asm volatile("s_wait_asynccnt 0x0" ::: "memory");
}

// ---------------- kernel 1: embed + concat + bf16 pack (K padded) ----------------
__global__ void k_embed(const int* __restrict__ nt_in, const int* __restrict__ t_in,
                        const float* __restrict__ nt_emb, const float* __restrict__ t_emb,
                        bf16* __restrict__ xb) {
  long idx = (long)blockIdx.x * blockDim.x + threadIdx.x;
  const long total = (long)SEQ * BATCH * KPAD;
  if (idx >= total) return;
  int k = (int)(idx % KPAD);
  long tb = idx / KPAD;              // t*BATCH + b
  float v = 0.0f;
  if (k < NT_DIM) {
    int tok = nt_in[tb];
    v = nt_emb[(long)tok * NT_DIM + k];
  } else if (k < IN_DIM) {
    int tok = t_in[tb];
    v = t_emb[(long)tok * T_DIM + (k - NT_DIM)];
  }
  xb[idx] = f2bf(v);
}

// ---------------- kernel 2: pack [W_ih | W_hh] -> column-major 32x16 tiles ----------------
// Bt layout: [nt (375)][kc (96)][nn (16)][kk (32)]  (bf16), kc<48 -> W_ih, else W_hh
__global__ void k_packW(const float* __restrict__ W_ih, const float* __restrict__ W_hh,
                        bf16* __restrict__ Bt) {
  long idx = (long)blockIdx.x * blockDim.x + threadIdx.x;
  const long total = (long)NT_TILES * KCH2 * 512;
  if (idx >= total) return;
  int e    = (int)(idx % 512);
  long tl  = idx / 512;
  int kc   = (int)(tl % KCH2);
  int nt   = (int)(tl / KCH2);
  int nn   = e >> 5;
  int kk   = e & 31;
  int n    = nt * 16 + nn;
  int kg   = kc * 32 + kk;
  float v = 0.0f;
  if (kg < KPAD) {
    if (kg < IN_DIM) v = W_ih[(long)n * IN_DIM + kg];
  } else {
    int k2 = kg - KPAD;
    if (k2 < HID) v = W_hh[(long)n * HID + k2];
  }
  Bt[idx] = f2bf(v);
}

// ---------------- kernel 3: pack W_out -> tiles [pnt(7)][kc(48)][16][32] ----------------
__global__ void k_packWout(const float* __restrict__ W_out, bf16* __restrict__ Bo) {
  long idx = (long)blockIdx.x * blockDim.x + threadIdx.x;
  const long total = (long)PNT * KCH * 512;
  if (idx >= total) return;
  int e   = (int)(idx % 512);
  long tl = idx / 512;
  int kc  = (int)(tl % KCH);
  int nt  = (int)(tl / KCH);
  int nn  = e >> 5;
  int kk  = e & 31;
  int n   = nt * 16 + nn;
  int kg  = kc * 32 + kk;
  float v = 0.0f;
  if (n < PRED && kg < HID) v = W_out[(long)n * HID + kg];
  Bo[idx] = f2bf(v);
}

// ---------------- kernel 4: init h,c (apply forget_vector), pack h -> bf16 ----------------
__global__ void k_init(const float* __restrict__ h0, const float* __restrict__ c0,
                       const float* __restrict__ fv,
                       float* __restrict__ h_cur, float* __restrict__ c_cur,
                       bf16* __restrict__ hb0) {
  int idx = blockIdx.x * blockDim.x + threadIdx.x;
  if (idx >= BATCH * KPAD) return;
  int b = idx / KPAD, j = idx % KPAD;
  if (j < HID) {
    float f = fv[b];
    float h = h0[b * HID + j] * f;
    float c = c0[b * HID + j] * f;
    h_cur[b * HID + j] = h;
    c_cur[b * HID + j] = c;
    hb0[idx] = f2bf(h);
  } else {
    hb0[idx] = f2bf(0.0f);
  }
}

// ---------------- kernel 5: per-step recurrent GEMM  gates = [x_t|h] @ [W_ih|W_hh]^T ----
// grid: 75 blocks x 256 thr (8 waves). wave = m-tile (0..7), block covers 5 n-tiles.
// B tiles staged into LDS via async DMA (double-buffered, 2 x 40KB); all 5 B
// fragments of a k-chunk are loaded into distinct registers before the 5 WMMAs
// so ds_load latency is pipelined instead of serialized per-WMMA.
__global__ void k_gemm_step(const bf16* __restrict__ xb_t,   // [128][1536]
                            const bf16* __restrict__ hprev,  // [128][1536]
                            const bf16* __restrict__ Bt,     // packed tiles
                            float* __restrict__ gates) {     // [128][6000]
  __shared__ bf16 smem[2][NPW][KSTAGE][512];   // 2 x 40KB

  const int tid  = threadIdx.x;
  const int wave = tid >> 5;
  const int lane = tid & 31;
  const int hi   = lane >> 4;
  const int lm   = lane & 15;
  const int m0   = wave * 16 + lm;
  const int nb   = blockIdx.x * NPW;

  // ---- stage loader: 40KB = 2560 x 16B units; 256 threads x 10 units ----
  auto issue_stage = [&](int s, int buf) {
    const int kc0 = s * KSTAGE;
#pragma unroll
    for (int it = 0; it < 10; ++it) {
      int c   = tid + it * 256;        // 16B unit index, 0..2559
      int ntl = c >> 9;                // n-tile segment (8KB each)
      int u   = c & 511;               // 16B unit within segment
      const char* g = (const char*)Bt
                    + (((size_t)(nb + ntl) * KCH2 + kc0) << 10) + ((size_t)u << 4);
      unsigned lds  = (unsigned)(size_t)&smem[buf][ntl][0][0] + (unsigned)(u << 4);
      async_copy_b128(lds, g);
    }
  };

  v8f acc[NPW] = {};
  issue_stage(0, 0);

#pragma unroll 1
  for (int s = 0; s < NSTAGE; ++s) {
    const int buf = s & 1;
    if (s + 1 < NSTAGE) {
      issue_stage(s + 1, buf ^ 1);
      wait_asynccnt_10();              // stage s resident (in-order completion)
    } else {
      wait_asynccnt_0();
    }
    __syncthreads();                   // all threads' portions visible

    // A base pointer is stage-uniform: stages 0..5 read x_t, 6..11 read h.
    const bf16* abase = (s < NSTAGE / 2)
        ? (xb_t  + (size_t)m0 * KPAD + (size_t)s * KSTAGE * 32)
        : (hprev + (size_t)m0 * KPAD + (size_t)(s - NSTAGE / 2) * KSTAGE * 32);

#pragma unroll
    for (int kcl = 0; kcl < KSTAGE; ++kcl) {
      v16bf a = load_frag32(abase + kcl * 32, hi);
      v16bf bfr[NPW];
#pragma unroll
      for (int j = 0; j < NPW; ++j)
        bfr[j] = load_frag32(&smem[buf][j][kcl][lm * 32], hi);
#pragma unroll
      for (int j = 0; j < NPW; ++j)
        acc[j] = __builtin_amdgcn_wmma_f32_16x16x32_bf16(
            false, a, false, bfr[j], (short)0, acc[j], false, false);
    }
    __syncthreads();                   // protect buf before next-stage overwrite
  }

  // C layout: vgpr i, lanes0-15 -> M=i, lanes16-31 -> M=i+8; N = lane&15
  const int mrow = wave * 16 + hi * 8;
#pragma unroll
  for (int j = 0; j < NPW; ++j) {
    int col = (nb + j) * 16 + lm;
#pragma unroll
    for (int i = 0; i < 8; ++i)
      gates[(size_t)(mrow + i) * G4 + col] = acc[j][i];
  }
}

// ---------------- kernel 6: LSTM cell elementwise + pack h (bf16, padded) ----------------
__global__ void k_cell(const float* __restrict__ gates,
                       const float* __restrict__ b_ih, const float* __restrict__ b_hh,
                       float* __restrict__ c_cur, float* __restrict__ h_cur,
                       bf16* __restrict__ hsb_t) {          // [128][1536]
  int idx = blockIdx.x * blockDim.x + threadIdx.x;
  if (idx >= BATCH * KPAD) return;
  int b = idx / KPAD, j = idx % KPAD;
  if (j >= HID) { hsb_t[idx] = f2bf(0.0f); return; }
  const float* g = gates + (size_t)b * G4;
  float ig = g[j]           + b_ih[j]           + b_hh[j];
  float fg = g[HID + j]     + b_ih[HID + j]     + b_hh[HID + j];
  float gg = g[2 * HID + j] + b_ih[2 * HID + j] + b_hh[2 * HID + j];
  float og = g[3 * HID + j] + b_ih[3 * HID + j] + b_hh[3 * HID + j];
  float c = sigm(fg) * c_cur[b * HID + j] + sigm(ig) * tanhf(gg);
  float h = sigm(og) * tanhf(c);
  c_cur[b * HID + j] = c;
  h_cur[b * HID + j] = h;
  hsb_t[idx] = f2bf(h);
}

// ---------------- kernel 7: output GEMM  logits = hs @ W_out^T (no bias) ----------------
// grid: 200 blocks x 8 waves; each wave: 1 m-tile (16 rows of 25600) x 7 n-tiles.
__global__ void k_gemm_out(const bf16* __restrict__ hsb,   // [25600][1536]
                           const bf16* __restrict__ Bo,    // packed [7][48][16][32]
                           float* __restrict__ logits) {   // [25600][112]
  const int wave = threadIdx.x >> 5;
  const int lane = threadIdx.x & 31;
  const int hi   = lane >> 4;
  const int lm   = lane & 15;
  const int mt   = blockIdx.x * 8 + wave;     // 0..1599
  const int m0   = mt * 16 + lm;

  v8f acc[PNT] = {};
#pragma unroll 1
  for (int kc = 0; kc < KCH; ++kc) {
    v16bf a = load_frag32(hsb + (size_t)m0 * KPAD + kc * 32, hi);
    v16bf bfr[PNT];
#pragma unroll
    for (int j = 0; j < PNT; ++j)
      bfr[j] = load_frag32(Bo + (((size_t)j * KCH + kc) * 512) + lm * 32, hi);
#pragma unroll
    for (int j = 0; j < PNT; ++j)
      acc[j] = __builtin_amdgcn_wmma_f32_16x16x32_bf16(
          false, a, false, bfr[j], (short)0, acc[j], false, false);
  }
  const int mrow = mt * 16 + hi * 8;
#pragma unroll
  for (int j = 0; j < PNT; ++j) {
    int col = j * 16 + lm;
#pragma unroll
    for (int i = 0; i < 8; ++i)
      logits[(size_t)(mrow + i) * PRED_PAD + col] = acc[j][i];
  }
}

// ---------------- kernel 8: log_softmax over 97 (wave per row) ----------------
__global__ void k_logsoftmax(const float* __restrict__ logits,
                             const float* __restrict__ b_out,
                             float* __restrict__ out) {     // [25600][97]
  const int wave = threadIdx.x >> 5;
  const int lane = threadIdx.x & 31;
  const int row  = blockIdx.x * 8 + wave;     // 0..25599
  const float* L = logits + (size_t)row * PRED_PAD;

  float v[4]; float mx = -3.0e38f;
#pragma unroll
  for (int s = 0; s < 4; ++s) {
    int p = lane + 32 * s;
    v[s] = (p < PRED) ? (L[p] + b_out[p]) : -3.0e38f;
    mx = fmaxf(mx, v[s]);
  }
#pragma unroll
  for (int d = 16; d >= 1; d >>= 1) mx = fmaxf(mx, __shfl_xor(mx, d, 32));
  float sum = 0.0f;
#pragma unroll
  for (int s = 0; s < 4; ++s) {
    int p = lane + 32 * s;
    if (p < PRED) sum += __expf(v[s] - mx);
  }
#pragma unroll
  for (int d = 16; d >= 1; d >>= 1) sum += __shfl_xor(sum, d, 32);
  float lse = __logf(sum);
#pragma unroll
  for (int s = 0; s < 4; ++s) {
    int p = lane + 32 * s;
    if (p < PRED) out[(size_t)row * PRED + p] = v[s] - mx - lse;
  }
}

// ---------------- kernel 9: copy final h, c into d_out ----------------
__global__ void k_copy_hc(const float* __restrict__ h_cur, const float* __restrict__ c_cur,
                          float* __restrict__ out_hc) {
  int idx = blockIdx.x * blockDim.x + threadIdx.x;
  if (idx >= BATCH * HID) return;
  out_hc[idx] = h_cur[idx];
  out_hc[BATCH * HID + idx] = c_cur[idx];
}

// ---------------- host-side launcher ----------------
extern "C" void kernel_launch(void* const* d_in, const int* in_sizes, int n_in,
                              void* d_out, int out_size, void* d_ws, size_t ws_size,
                              hipStream_t stream) {
  const int*   nt_in  = (const int*)d_in[0];
  const int*   t_in   = (const int*)d_in[1];
  const float* h0     = (const float*)d_in[2];
  const float* c0     = (const float*)d_in[3];
  const float* fv     = (const float*)d_in[4];
  const float* nt_emb = (const float*)d_in[5];
  const float* t_emb  = (const float*)d_in[6];
  const float* W_ih   = (const float*)d_in[7];
  const float* W_hh   = (const float*)d_in[8];
  const float* b_ih   = (const float*)d_in[9];
  const float* b_hh   = (const float*)d_in[10];
  const float* W_out  = (const float*)d_in[11];
  const float* b_out  = (const float*)d_in[12];
  float* out = (float*)d_out;

  // workspace carve-up (256B aligned)
  char* ws = (char*)d_ws;
  auto take = [&](size_t bytes) { char* p = ws; ws += (bytes + 255) & ~(size_t)255; return p; };
  bf16*  xb     = (bf16*)take((size_t)SEQ * BATCH * KPAD * 2);   // 78.6 MB
  bf16*  hsb    = (bf16*)take((size_t)SEQ * BATCH * KPAD * 2);   // 78.6 MB
  bf16*  Bt     = (bf16*)take((size_t)NT_TILES * KCH2 * 512 * 2);// 36.9 MB
  bf16*  Bo     = (bf16*)take((size_t)PNT * KCH * 512 * 2);      // 0.34 MB
  bf16*  hb0    = (bf16*)take((size_t)BATCH * KPAD * 2);
  float* h_cur  = (float*)take((size_t)BATCH * HID * 4);
  float* c_cur  = (float*)take((size_t)BATCH * HID * 4);
  float* gates  = (float*)take((size_t)BATCH * G4 * 4);          // 3.1 MB
  float* logits = (float*)take((size_t)MROWS * PRED_PAD * 4);    // 11.5 MB

  const int B256 = 256;
  // prep
  {
    long n = (long)SEQ * BATCH * KPAD;
    k_embed<<<(unsigned)((n + B256 - 1) / B256), B256, 0, stream>>>(nt_in, t_in, nt_emb, t_emb, xb);
  }
  {
    long n = (long)NT_TILES * KCH2 * 512;
    k_packW<<<(unsigned)((n + B256 - 1) / B256), B256, 0, stream>>>(W_ih, W_hh, Bt);
  }
  {
    long n = (long)PNT * KCH * 512;
    k_packWout<<<(unsigned)((n + B256 - 1) / B256), B256, 0, stream>>>(W_out, Bo);
  }
  k_init<<<(BATCH * KPAD + B256 - 1) / B256, B256, 0, stream>>>(h0, c0, fv, h_cur, c_cur, hb0);

  // sequential LSTM
  for (int t = 0; t < SEQ; ++t) {
    const bf16* xt    = xb  + (size_t)t * BATCH * KPAD;
    const bf16* hprev = (t == 0) ? hb0 : (hsb + (size_t)(t - 1) * BATCH * KPAD);
    bf16*       ht    = hsb + (size_t)t * BATCH * KPAD;
    k_gemm_step<<<NT_TILES / NPW, 256, 0, stream>>>(xt, hprev, Bt, gates);
    k_cell<<<(BATCH * KPAD + B256 - 1) / B256, B256, 0, stream>>>(gates, b_ih, b_hh, c_cur, h_cur, ht);
  }

  // projection + log_softmax
  k_gemm_out<<<MROWS / 16 / 8, 256, 0, stream>>>(hsb, Bo, logits);
  k_logsoftmax<<<MROWS / 8, 256, 0, stream>>>(logits, b_out, out);
  k_copy_hc<<<(BATCH * HID + B256 - 1) / B256, B256, 0, stream>>>(h_cur, c_cur,
                                                                 out + (size_t)SEQ * BATCH * PRED);
}
